// Decoder_67482526155227
// MI455X (gfx1250) — matrix operand order repaired and checked by reading, not compiled, not run on hardware
//
#include <hip/hip_runtime.h>

// Problem constants (match reference)
#define LL 2
#define EE 1024
#define HH 1024
#define SS 128
#define TT 64
#define BB 128

typedef __attribute__((ext_vector_type(16))) __bf16 v16bf;
typedef __attribute__((ext_vector_type(8)))  __bf16 v8bf;
typedef __attribute__((ext_vector_type(8)))  float  v8f;

__device__ __forceinline__ float fsig(float x) {
  return 1.0f / (1.0f + __expf(-x));
}
__device__ __forceinline__ float ftanh(float x) {
  x = fminf(fmaxf(x, -15.0f), 15.0f);
  float e = __expf(2.0f * x);
  return (e - 1.0f) / (e + 1.0f);
}

// ---------------------------------------------------------------------------
// Prep kernels: pack weights to bf16 layouts consumed by the WMMA GEMM.
// ---------------------------------------------------------------------------

// Wg[l][n][k] bf16, k<1024 -> w_ih[l][n][k], else w_hh[l][n][k-1024].
__global__ void pack_gates_w_kernel(const float* __restrict__ wih,
                                    const float* __restrict__ whh,
                                    __bf16* __restrict__ Wg) {
  size_t idx = (size_t)blockIdx.x * 256 + threadIdx.x;  // over LL*4H*2H
  int k = (int)(idx & (2 * HH - 1));
  size_t nl = idx >> 11;  // l*4H + n
  float v = (k < HH) ? wih[nl * EE + k] : whh[nl * HH + (k - HH)];
  Wg[idx] = (__bf16)v;
}

__global__ void cvt_bf16_kernel(const float* __restrict__ src,
                                __bf16* __restrict__ dst) {
  size_t idx = (size_t)blockIdx.x * 256 + threadIdx.x;
  dst[idx] = (__bf16)src[idx];
}

__global__ void pack_bias_kernel(const float* __restrict__ bih,
                                 const float* __restrict__ bhh,
                                 float* __restrict__ bg) {
  int idx = blockIdx.x * 256 + threadIdx.x;  // LL*4H
  bg[idx] = bih[idx] + bhh[idx];
}

// Gather all T*B embeddings to bf16 once (PAD row is already zero).
__global__ void embed_kernel(const int* __restrict__ inp,
                             const float* __restrict__ embW,
                             __bf16* __restrict__ xemb) {
  size_t idx = (size_t)blockIdx.x * 256 + threadIdx.x;  // T*B*E
  int e = (int)(idx & (EE - 1));
  size_t tb = idx >> 10;
  int tok = inp[tb];
  xemb[idx] = (__bf16)embW[(size_t)tok * EE + e];
}

__global__ void init_state_kernel(const float* __restrict__ h0,
                                  const float* __restrict__ c0,
                                  float* __restrict__ h_st,
                                  float* __restrict__ c_st,
                                  __bf16* __restrict__ hbf) {
  int idx = blockIdx.x * 256 + threadIdx.x;  // L*B*H
  float hv = h0[idx];
  h_st[idx] = hv;
  c_st[idx] = c0[idx];
  hbf[idx] = (__bf16)hv;
}

// ---------------------------------------------------------------------------
// WMMA GEMM: C[M=128, N] = A[M, Ktot] * W^T (+bias)(+tanh), fp32 accumulate.
// A is split: k < ksplit reads Alo, else Ahi (fuses x@Wih^T + h@Whh^T).
// The split is handled as two sequential K-phases with a fixed A base per
// phase (phase-2 base biased by -ksplit), so the inner loop has no pointer
// selects -- just base+kk addressing folded into load offsets.
// W is row-major [N, Ktot] bf16 (row n = output column n, contiguous in K).
//
// Register blocking: block = 128 threads (4 wave32). Each wave owns a
// 32(M) x 64(N) C tile = 2 A fragments x 4 B fragments -> 8 accumulators,
// 8 WMMAs per K-step against 12 b128 loads (1.5 loads/WMMA).
// Block covers 32(M) x 256(N); grid = (N/256, M/32).
//
// Fragment layouts used here (wave32; A/B k order is this kernel's contiguous choice, not ISA table order):
//   A 16x32 bf16: lane m = lane&15, kb = 16*(lane>>4); elements e hold
//                 K=kb+e (16 contiguous bf16, matching B).
//   B 32x16 bf16: lane col n = lane&15, kh = 16*(lane>>4); elements e hold
//                 K=kh+e (16 contiguous bf16 from weight row n).
//   C 16x16 f32:  elem j -> row 8*(lane>>4)+j, col lane&15.
// EPI: 0 = +bias (LSTM gates), 1 = raw (q projection), 2 = tanh (output).
// ---------------------------------------------------------------------------
template <int EPI>
__global__ __launch_bounds__(128) void wmma_gemm_kernel(
    const __bf16* __restrict__ Alo, const __bf16* __restrict__ Ahi,
    int ksplit, int ktot, int lda_lo, int lda_hi,
    const __bf16* __restrict__ W, int ldw,
    const float* __restrict__ bias,
    float* __restrict__ C, int ldc) {
  const int lane = threadIdx.x & 31;
  const int wave = threadIdx.x >> 5;
  const int m0 = blockIdx.y * 32;
  const int nb = blockIdx.x * 256 + wave * 64;

  const int am  = lane & 15;
  const int akb = (lane >> 4) << 4;
  const int bn  = lane & 15;
  const int bkh = (lane >> 4) << 4;

  const __bf16* wrow[4];
#pragma unroll
  for (int nj = 0; nj < 4; ++nj) {
    wrow[nj] = W + (size_t)(nb + 16 * nj + bn) * ldw + bkh;
  }

  v8f acc[2][4];
#pragma unroll
  for (int mi = 0; mi < 2; ++mi)
#pragma unroll
    for (int nj = 0; nj < 4; ++nj) acc[mi][nj] = (v8f){};

  // One K-phase: A base fixed, k indexes [kbeg, kend) directly (caller biases
  // the base pointer so the global k works for the phase-2 matrix too).
  auto run_phase = [&](const __bf16* A, int lda, int kofs, int kbeg,
                       int kend) {
    const __bf16* ar0 = A + (size_t)(m0 + am) * lda + akb - kofs;
    const __bf16* ar1 = A + (size_t)(m0 + 16 + am) * lda + akb - kofs;
#pragma unroll 2
    for (int kk = kbeg; kk < kend; kk += 32) {
      v8bf a0lo = *(const v8bf*)(ar0 + kk);
      v8bf a0hi = *(const v8bf*)(ar0 + kk + 8);
      v8bf a1lo = *(const v8bf*)(ar1 + kk);
      v8bf a1hi = *(const v8bf*)(ar1 + kk + 8);
      v16bf af0 = __builtin_shufflevector(a0lo, a0hi, 0, 1, 2, 3, 4, 5, 6, 7,
                                          8, 9, 10, 11, 12, 13, 14, 15);
      v16bf af1 = __builtin_shufflevector(a1lo, a1hi, 0, 1, 2, 3, 4, 5, 6, 7,
                                          8, 9, 10, 11, 12, 13, 14, 15);
#pragma unroll
      for (int nj = 0; nj < 4; ++nj) {
        v16bf b = *(const v16bf*)(wrow[nj] + kk);
        acc[0][nj] = __builtin_amdgcn_wmma_f32_16x16x32_bf16(
            false, af0, false, b, (short)0, acc[0][nj], false, false);
        acc[1][nj] = __builtin_amdgcn_wmma_f32_16x16x32_bf16(
            false, af1, false, b, (short)0, acc[1][nj], false, false);
      }
    }
  };

  run_phase(Alo, lda_lo, 0, 0, ksplit);
  if (ksplit < ktot) run_phase(Ahi, lda_hi, ksplit, ksplit, ktot);

  const int cr = (lane >> 4) << 3;
  const int cc = lane & 15;
#pragma unroll
  for (int mi = 0; mi < 2; ++mi) {
#pragma unroll
    for (int nj = 0; nj < 4; ++nj) {
      const int col = nb + 16 * nj + cc;
#pragma unroll
      for (int j = 0; j < 8; ++j) {
        float v = acc[mi][nj][j];
        if (EPI == 0) v += bias[col];
        if (EPI == 2) v = ftanh(v);
        C[(size_t)(m0 + 16 * mi + cr + j) * ldc + col] = v;
      }
    }
  }
}

// ---------------------------------------------------------------------------
// LSTM gate nonlinearity: c,h update in fp32; also emit bf16 h for the next
// GEMM, and optionally into catbuf[:,1024:2048] (layer-1 h feeds attention).
// ---------------------------------------------------------------------------
__global__ void lstm_cell_kernel(const float* __restrict__ gates,
                                 float* __restrict__ c_st,
                                 float* __restrict__ h_st,
                                 __bf16* __restrict__ h_bf,
                                 __bf16* __restrict__ cat_h) {
  int idx = blockIdx.x * 256 + threadIdx.x;  // B*H
  int b = idx >> 10;
  int hh = idx & (HH - 1);
  const float* g = gates + (size_t)b * 4 * HH;
  float ig = g[hh];
  float fg = g[HH + hh];
  float gg = g[2 * HH + hh];
  float og = g[3 * HH + hh];
  float c = c_st[idx];
  float cn = fsig(fg) * c + fsig(ig) * ftanh(gg);
  float hn = fsig(og) * ftanh(cn);
  c_st[idx] = cn;
  h_st[idx] = hn;
  h_bf[idx] = (__bf16)hn;
  if (cat_h) cat_h[(size_t)b * 2 * HH + HH + hh] = (__bf16)hn;
}

// ---------------------------------------------------------------------------
// Attention: scores = q . ctx, softmax over S (LDS), wc = a @ ctx.
// One block per batch row, 128 threads. a written straight into d_out's attn
// slot (last timestep's write survives). wc -> bf16 catbuf[:,0:1024].
// ---------------------------------------------------------------------------
__global__ __launch_bounds__(SS) void attn_kernel(
    const float* __restrict__ q, const float* __restrict__ ctx,
    float* __restrict__ attn_out, __bf16* __restrict__ catb) {
  __shared__ float s_sc[SS];
  int b = blockIdx.x;
  int tid = threadIdx.x;
  const float* qb = q + (size_t)b * HH;
  const float* cb = ctx + (size_t)b * SS * HH;

  // score for source position s = tid
  const float* crow = cb + (size_t)tid * HH;
  float acc = 0.0f;
  for (int d = 0; d < HH; d += 4) {
    acc += qb[d] * crow[d] + qb[d + 1] * crow[d + 1] +
           qb[d + 2] * crow[d + 2] + qb[d + 3] * crow[d + 3];
  }
  s_sc[tid] = acc;
  __syncthreads();
  float mx = -3.4e38f;
  for (int i = 0; i < SS; ++i) mx = fmaxf(mx, s_sc[i]);
  float ex = __expf(acc - mx);
  __syncthreads();
  s_sc[tid] = ex;
  __syncthreads();
  float sum = 0.0f;
  for (int i = 0; i < SS; ++i) sum += s_sc[i];
  float a = ex / sum;
  attn_out[(size_t)b * SS + tid] = a;
  __syncthreads();
  s_sc[tid] = a;
  __syncthreads();

  // weighted context
  for (int d = tid; d < HH; d += SS) {
    float wv = 0.0f;
    for (int s2 = 0; s2 < SS; ++s2) wv += s_sc[s2] * cb[(size_t)s2 * HH + d];
    catb[(size_t)b * 2 * HH + d] = (__bf16)wv;
  }
}

// ---------------------------------------------------------------------------
// Host orchestration
// ---------------------------------------------------------------------------
extern "C" void kernel_launch(void* const* d_in, const int* in_sizes, int n_in,
                              void* d_out, int out_size, void* d_ws,
                              size_t ws_size, hipStream_t stream) {
  (void)in_sizes; (void)n_in; (void)out_size; (void)ws_size;
  const int*   inp  = (const int*)d_in[0];
  const float* h0   = (const float*)d_in[1];
  const float* c0   = (const float*)d_in[2];
  const float* ctx  = (const float*)d_in[3];
  // d_in[4] init_output: unused (input_feed=0 path)
  const float* embW = (const float*)d_in[5];
  const float* wih  = (const float*)d_in[6];
  const float* whh  = (const float*)d_in[7];
  const float* bih  = (const float*)d_in[8];
  const float* bhh  = (const float*)d_in[9];
  const float* Win  = (const float*)d_in[10];
  const float* Wout = (const float*)d_in[11];

  float* out_outputs = (float*)d_out;                       // [T,B,H]
  float* out_h = out_outputs + (size_t)TT * BB * HH;        // [L,B,H]
  float* out_c = out_h + (size_t)LL * BB * HH;              // [L,B,H]
  float* out_attn = out_c + (size_t)LL * BB * HH;           // [B,S]

  // workspace carve (all sizes are multiples of 256B; hipMalloc base aligned)
  char* p = (char*)d_ws;
  auto take = [&](size_t bytes) {
    char* r = p;
    p += (bytes + 255) & ~(size_t)255;
    return r;
  };
  __bf16* Wg    = (__bf16*)take((size_t)LL * 4 * HH * 2 * HH * sizeof(__bf16));
  __bf16* Winb  = (__bf16*)take((size_t)HH * HH * sizeof(__bf16));
  __bf16* Woutb = (__bf16*)take((size_t)HH * 2 * HH * sizeof(__bf16));
  float*  bg    = (float*)take((size_t)LL * 4 * HH * sizeof(float));
  __bf16* xemb  = (__bf16*)take((size_t)TT * BB * EE * sizeof(__bf16));
  __bf16* hbf   = (__bf16*)take((size_t)LL * BB * HH * sizeof(__bf16));
  float*  gates = (float*)take((size_t)BB * 4 * HH * sizeof(float));
  float*  qbuf  = (float*)take((size_t)BB * HH * sizeof(float));
  __bf16* catb  = (__bf16*)take((size_t)BB * 2 * HH * sizeof(__bf16));

  // ---- prep (independent, re-done every call for determinism) ----
  pack_gates_w_kernel<<<(unsigned)((size_t)LL * 4 * HH * 2 * HH / 256), 256, 0,
                        stream>>>(wih, whh, Wg);
  cvt_bf16_kernel<<<(HH * HH) / 256, 256, 0, stream>>>(Win, Winb);
  cvt_bf16_kernel<<<(HH * 2 * HH) / 256, 256, 0, stream>>>(Wout, Woutb);
  pack_bias_kernel<<<(LL * 4 * HH) / 256, 256, 0, stream>>>(bih, bhh, bg);
  embed_kernel<<<(unsigned)((size_t)TT * BB * EE / 256), 256, 0, stream>>>(
      inp, embW, xemb);
  init_state_kernel<<<(LL * BB * HH) / 256, 256, 0, stream>>>(h0, c0, out_h,
                                                              out_c, hbf);

  const dim3 gatesGrid(4 * HH / 256, BB / 32);  // (16, 4)
  const dim3 projGrid(HH / 256, BB / 32);       // (4, 4)

  // ---- serial timestep loop ----
  for (int t = 0; t < TT; ++t) {
    const __bf16* xt = xemb + (size_t)t * BB * EE;

    // layer 0: gates = [x_emb | h0] @ [Wih|Whh]^T + b
    wmma_gemm_kernel<0><<<gatesGrid, 128, 0, stream>>>(
        xt, hbf, EE, EE + HH, EE, HH, Wg, 2 * HH, bg, gates, 4 * HH);
    lstm_cell_kernel<<<(BB * HH) / 256, 256, 0, stream>>>(
        gates, out_c, out_h, hbf, (__bf16*)nullptr);

    // layer 1: gates = [h0_new | h1] @ [Wih|Whh]^T + b
    wmma_gemm_kernel<0><<<gatesGrid, 128, 0, stream>>>(
        hbf, hbf + (size_t)BB * HH, HH, 2 * HH, HH, HH,
        Wg + (size_t)4 * HH * 2 * HH, 2 * HH, bg + 4 * HH, gates, 4 * HH);
    lstm_cell_kernel<<<(BB * HH) / 256, 256, 0, stream>>>(
        gates, out_c + (size_t)BB * HH, out_h + (size_t)BB * HH,
        hbf + (size_t)BB * HH, catb);

    // attention: q = h1 @ W_in^T ; softmax(q.ctx) ; wc
    wmma_gemm_kernel<1><<<projGrid, 128, 0, stream>>>(
        hbf + (size_t)BB * HH, hbf + (size_t)BB * HH, HH, HH, HH, HH, Winb, HH,
        (const float*)nullptr, qbuf, HH);
    attn_kernel<<<BB, SS, 0, stream>>>(qbuf, ctx, out_attn, catb);

    // out[t] = tanh([wc | h1] @ W_out^T)
    wmma_gemm_kernel<2><<<projGrid, 128, 0, stream>>>(
        catb, catb, 2 * HH, 2 * HH, 2 * HH, 2 * HH, Woutb, 2 * HH,
        (const float*)nullptr, out_outputs + (size_t)t * BB * HH, HH);
  }
}
